// LinearAttention_60919816126617
// MI455X (gfx1250) — matrix operand order repaired and checked by reading, not compiled, not run on hardware
//
#include <hip/hip_runtime.h>
#include <hip/hip_bf16.h>

typedef __bf16 bf16;
typedef __attribute__((ext_vector_type(8)))  __bf16 bf16x8;
typedef __attribute__((ext_vector_type(16))) __bf16 bf16x16;
typedef __attribute__((ext_vector_type(8)))  float  f32x8;

// ---------------------------------------------------------------------------
// WMMA fragment helpers (CDNA5 ISA 7.12.2 layouts, wave32)
// A 16x32 bf16:  lane<16 row m=lane, holds K = {0..7, 16..23}
//                lane>=16 row m=lane-16, holds K = {8..15, 24..31}
// B 32x16 bf16:  lane<16 col n=lane, holds K = {0..7, 16..23} ; lane>=16 K = {8..15, 24..31}
// C/D 16x16 f32: VGPR r -> row r + 8*(lane>=16), col = lane&15
// ---------------------------------------------------------------------------
__device__ __forceinline__ bf16x16 cat8(bf16x8 lo, bf16x8 hi) {
  return __builtin_shufflevector(lo, hi, 0,1,2,3,4,5,6,7,8,9,10,11,12,13,14,15);
}

// rowBase: 32 contiguous bf16 K-values for row m = lane&15
__device__ __forceinline__ bf16x16 load_a_frag(const bf16* rowBase, int lane) {
  const int k0 = (lane < 16) ? 0 : 8;
  bf16x8 lo = *(const bf16x8*)(rowBase + k0);
  bf16x8 hi = *(const bf16x8*)(rowBase + k0 + 16);
  return cat8(lo, hi);
}

// rowBase: 32 contiguous bf16 K-values for col n = lane&15 (B^T row)
__device__ __forceinline__ bf16x16 load_b_frag(const bf16* rowBase, int lane) {
  const int k0 = (lane < 16) ? 0 : 8;
  bf16x8 lo = *(const bf16x8*)(rowBase + k0);
  bf16x8 hi = *(const bf16x8*)(rowBase + k0 + 16);
  return cat8(lo, hi);
}

// ---------------------------------------------------------------------------
// CDNA5 async copy: global -> LDS, 16 bytes per lane, tracked by ASYNCcnt.
// LDS generic addresses carry the LDS byte offset in the low 32 bits.
// ---------------------------------------------------------------------------
__device__ __forceinline__ void async_copy_b128(void* lds_ptr, const void* gptr) {
  unsigned int loff = (unsigned int)(uintptr_t)lds_ptr;
  asm volatile("global_load_async_to_lds_b128 %0, %1, off"
               :: "v"(loff), "v"(gptr) : "memory");
}
__device__ __forceinline__ void async_wait3() {
  asm volatile("s_wait_asynccnt 0x3" ::: "memory");
}
__device__ __forceinline__ void async_wait0() {
  asm volatile("s_wait_asynccnt 0x0" ::: "memory");
}

// ---------------------------------------------------------------------------
// Kernel 0: fp32 -> bf16 weight conversion (run once per weight matrix)
// ---------------------------------------------------------------------------
__global__ __launch_bounds__(256)
void f32_to_bf16_kernel(const float* __restrict__ src, bf16* __restrict__ dst, int n)
{
  int i = blockIdx.x * 256 + threadIdx.x;
  if (i < n) dst[i] = (bf16)src[i];
}

// ---------------------------------------------------------------------------
// Kernel 1: channel LayerNorm over c=512 + convert to bf16, TRANSPOSED
// output layout Xt[b][n][c] so GEMM-B fragments are contiguous-K rows.
// ---------------------------------------------------------------------------
__global__ __launch_bounds__(256)
void ln_bf16_t_kernel(const float* __restrict__ x, const float* __restrict__ g,
                      bf16* __restrict__ xt)
{
  const int n = blockIdx.x * 256 + threadIdx.x;          // 0..4095
  const int b = blockIdx.y;
  const float* xp = x + (long)b * 512 * 4096 + n;
  float s = 0.f, ss = 0.f;
  for (int c = 0; c < 512; ++c) {
    float v = xp[(long)c * 4096];
    s += v; ss += v * v;
  }
  const float mean = s * (1.0f / 512.0f);
  const float rstd = rsqrtf(ss * (1.0f / 512.0f) - mean * mean + 1e-5f);
  bf16* op = xt + ((long)b * 4096 + n) * 512;
  for (int c0 = 0; c0 < 512; c0 += 8) {
    bf16x8 pk;
#pragma unroll
    for (int j = 0; j < 8; ++j)
      pk[j] = (bf16)((xp[(long)(c0 + j) * 4096] - mean) * rstd * g[c0 + j]);
    *(bf16x8*)(op + c0) = pk;                            // b128 store
  }
}

// ---------------------------------------------------------------------------
// bf16 WMMA GEMM: Out[b] = Wb (MxK bf16) @ Xt[b]^T  (Xt is NxK bf16 row-major)
// Block tile 128x64, 8 waves x (2x2) WMMA tiles, TK=32.
// Both LDS tiles are staged with global_load_async_to_lds_b128, double-buffered
// and overlapped with WMMA via s_wait_asynccnt.
// ---------------------------------------------------------------------------
template<bool OUT_BF16>
__global__ __launch_bounds__(256)
void gemm_wmma_kernel(const bf16* __restrict__ Wb, const bf16* __restrict__ Xt,
                      void* __restrict__ Out, int M, int K, int N,
                      long xBatchStride, long oBatchStride)
{
  constexpr int TM = 128, TN = 64, TK = 32;
  __shared__ __align__(16) bf16 lA[2][TM][TK];   // [m][k]
  __shared__ __align__(16) bf16 lB[2][TN][TK];   // [n][k] (already B^T)

  const int tid = threadIdx.x;
  const int n0  = blockIdx.x * TN;
  const int m0  = blockIdx.y * TM;
  const int b   = blockIdx.z;
  const bf16* Xb = Xt + (long)b * xBatchStride;

  const int wave = tid >> 5, lane = tid & 31;
  const int wm = (wave & 3) * 32;
  const int wn = (wave >> 2) * 32;

  // async stage of one K-tile: A = 2 chunks/thread, B = 1 chunk/thread (16B each)
  auto stage = [&](int buf, int kk) {
    const int c0 = tid, c1 = tid + 256;
    async_copy_b128(&lA[buf][c0 >> 2][(c0 & 3) * 8],
                    Wb + (long)(m0 + (c0 >> 2)) * K + kk + (c0 & 3) * 8);
    async_copy_b128(&lA[buf][c1 >> 2][(c1 & 3) * 8],
                    Wb + (long)(m0 + (c1 >> 2)) * K + kk + (c1 & 3) * 8);
    async_copy_b128(&lB[buf][tid >> 2][(tid & 3) * 8],
                    Xb + (long)(n0 + (tid >> 2)) * K + kk + (tid & 3) * 8);
  };

  f32x8 acc[2][2] = {};

  const int NK = K / TK;
  stage(0, 0);
  for (int t = 0; t < NK; ++t) {
    const int cur = t & 1;
    if (t + 1 < NK) { stage(1 - cur, (t + 1) * TK); async_wait3(); }
    else            { async_wait0(); }
    __syncthreads();

#pragma unroll
    for (int mi = 0; mi < 2; ++mi) {
      bf16x16 afrag = load_a_frag(&lA[cur][wm + mi * 16 + (lane & 15)][0], lane);
#pragma unroll
      for (int ni = 0; ni < 2; ++ni) {
        bf16x16 bfrag = load_b_frag(&lB[cur][wn + ni * 16 + (lane & 15)][0], lane);
        acc[mi][ni] = __builtin_amdgcn_wmma_f32_16x16x32_bf16(
            false, afrag, false, bfrag, (short)0, acc[mi][ni], false, false);
      }
    }
    __syncthreads();
  }

  const int colLane = lane & 15;
  const int rowOff  = (lane >> 4) * 8;
#pragma unroll
  for (int mi = 0; mi < 2; ++mi)
#pragma unroll
    for (int ni = 0; ni < 2; ++ni)
#pragma unroll
      for (int r = 0; r < 8; ++r) {
        const int row = m0 + wm + mi * 16 + rowOff + r;
        const int col = n0 + wn + ni * 16 + colLane;
        const float v = acc[mi][ni][r];
        if (OUT_BF16)
          ((bf16*)Out)[(long)b * oBatchStride + (long)row * N + col] = (bf16)v;
        else
          ((float*)Out)[(long)b * oBatchStride + (long)row * N + col] = v;
      }
}

// ---------------------------------------------------------------------------
// Kernel 3: q softmax over feature dim (64) per (b,h,n), *dh^-0.5, in place
// qkv layout: [b][o][n], o = section*512 + h*64 + d
// ---------------------------------------------------------------------------
__global__ __launch_bounds__(256)
void q_softmax_kernel(bf16* __restrict__ qkv)
{
  const int n = blockIdx.x * 256 + threadIdx.x;
  const int h = blockIdx.y, b = blockIdx.z;
  bf16* qp = qkv + ((long)(b * 1536 + h * 64)) * 4096 + n;
  float v[64];
  float mx = -3.0e38f;
#pragma unroll
  for (int d = 0; d < 64; ++d) { v[d] = (float)qp[(long)d * 4096]; mx = fmaxf(mx, v[d]); }
  float sum = 0.f;
#pragma unroll
  for (int d = 0; d < 64; ++d) { v[d] = __expf(v[d] - mx); sum += v[d]; }
  const float sc = 0.125f / sum;   // softmax * dh^-0.5
#pragma unroll
  for (int d = 0; d < 64; ++d) qp[(long)d * 4096] = (bf16)(v[d] * sc);
}

// ---------------------------------------------------------------------------
// Kernel 4: k softmax over seq dim (4096) per (b,h,d), * 1/sqrt(n), in place
// ---------------------------------------------------------------------------
__global__ __launch_bounds__(256)
void k_softmax_kernel(bf16* __restrict__ qkv)
{
  __shared__ float red[256];
  const int tid = threadIdx.x;
  const int o = blockIdx.x;            // b*512 + (h*64+d)
  const int b = o >> 9, r = o & 511;
  bf16* kp = qkv + ((long)(b * 1536 + 512 + r)) * 4096;
  float v[16];
  float mx = -3.0e38f;
#pragma unroll
  for (int i = 0; i < 16; ++i) { v[i] = (float)kp[tid + i * 256]; mx = fmaxf(mx, v[i]); }
  red[tid] = mx; __syncthreads();
  for (int s = 128; s > 0; s >>= 1) { if (tid < s) red[tid] = fmaxf(red[tid], red[tid + s]); __syncthreads(); }
  mx = red[0]; __syncthreads();
  float sum = 0.f;
#pragma unroll
  for (int i = 0; i < 16; ++i) { v[i] = __expf(v[i] - mx); sum += v[i]; }
  red[tid] = sum; __syncthreads();
  for (int s = 128; s > 0; s >>= 1) { if (tid < s) red[tid] += red[tid + s]; __syncthreads(); }
  const float sc = 0.015625f / red[0];   // softmax * 1/sqrt(4096)
#pragma unroll
  for (int i = 0; i < 16; ++i) kp[tid + i * 256] = (bf16)(v[i] * sc);
}

// ---------------------------------------------------------------------------
// Kernel 5: v l2norm over feature dim, * 1/sqrt(n), in place
// ---------------------------------------------------------------------------
__global__ __launch_bounds__(256)
void v_l2norm_kernel(bf16* __restrict__ qkv)
{
  const int n = blockIdx.x * 256 + threadIdx.x;
  const int h = blockIdx.y, b = blockIdx.z;
  bf16* vp = qkv + ((long)(b * 1536 + 1024 + h * 64)) * 4096 + n;
  float v[64];
  float ss = 0.f;
#pragma unroll
  for (int d = 0; d < 64; ++d) { v[d] = (float)vp[(long)d * 4096]; ss += v[d] * v[d]; }
  const float sc = rsqrtf(ss) * 0.015625f;
#pragma unroll
  for (int d = 0; d < 64; ++d) vp[(long)d * 4096] = (bf16)(v[d] * sc);
}

// ---------------------------------------------------------------------------
// Kernel 6: context^T[b][h][e][d] = sum_n v'[e][n] k'[d][n]  (bf16 WMMA, K=4096)
// Operands are [d][n] row-major -> fragments load straight from global.
// 512 threads = 16 waves, one 16x16 tile each.
// ---------------------------------------------------------------------------
__global__ __launch_bounds__(512)
void context_kernel(const bf16* __restrict__ qkv, bf16* __restrict__ ctxT)
{
  const int bh = blockIdx.x;           // 0..127
  const int b = bh >> 3, h = bh & 7;
  const int tid = threadIdx.x;
  const int wave = tid >> 5, lane = tid & 31;
  const int m16 = (wave >> 2) * 16;    // d block
  const int n16 = (wave & 3) * 16;     // e block

  const bf16* krow = qkv + ((long)(b * 1536 + 512  + h * 64 + m16 + (lane & 15))) * 4096;
  const bf16* vrow = qkv + ((long)(b * 1536 + 1024 + h * 64 + n16 + (lane & 15))) * 4096;

  f32x8 acc = {};
  for (int kk = 0; kk < 4096; kk += 32) {
    bf16x16 a  = load_a_frag(krow + kk, lane);
    bf16x16 bm = load_b_frag(vrow + kk, lane);
    acc = __builtin_amdgcn_wmma_f32_16x16x32_bf16(false, a, false, bm, (short)0, acc, false, false);
  }
  bf16* ct = ctxT + (long)bh * 64 * 64;  // [e][d]
  const int rowOff = (lane >> 4) * 8;
#pragma unroll
  for (int r = 0; r < 8; ++r) {
    const int d = m16 + rowOff + r;
    const int e = n16 + (lane & 15);
    ct[e * 64 + d] = (bf16)acc[r];       // transposed store
  }
}

// ---------------------------------------------------------------------------
// Kernel 7: attn[b][n][h*64+e] = GELU( sum_d q'[d][n] * ctxT[e][d] )
// A = q tile transposed through LDS (rows n, contiguous d),
// B = ctxT rows (cols e, contiguous d) straight from global.
// Output is [n][c] so the final GEMM consumes it as B^T with contiguous K.
// ---------------------------------------------------------------------------
__global__ __launch_bounds__(256)
void attn_out_kernel(const bf16* __restrict__ qkv, const bf16* __restrict__ ctxT,
                     bf16* __restrict__ attn)
{
  const int n0 = blockIdx.x * 64;
  const int bh = blockIdx.y;
  const int b = bh >> 3, h = bh & 7;
  const int tid = threadIdx.x, wave = tid >> 5, lane = tid & 31;

  __shared__ __align__(16) bf16 lQt[64][64];   // [n][d]

  const bf16* qbase = qkv + ((long)(b * 1536 + h * 64)) * 4096;
#pragma unroll
  for (int i = 0; i < 16; ++i) {
    int idx = tid + i * 256;
    int d = idx >> 6, n = idx & 63;
    lQt[n][d] = qbase[(long)d * 4096 + n0 + n];
  }
  __syncthreads();

  const int m16   = (wave & 3) * 16;    // n block (M dim)
  const int nbase = (wave >> 2) * 32;   // two 16-wide e tiles per wave
  const bf16* ct = ctxT + (long)bh * 4096;

#pragma unroll
  for (int nt = 0; nt < 2; ++nt) {
    const int n16 = nbase + nt * 16;    // e block (N dim)
    f32x8 acc = {};
#pragma unroll
    for (int kk = 0; kk < 64; kk += 32) {
      bf16x16 a  = load_a_frag(&lQt[m16 + (lane & 15)][kk], lane);       // row n
      bf16x16 bm = load_b_frag(ct + (n16 + (lane & 15)) * 64 + kk, lane); // col e
      acc = __builtin_amdgcn_wmma_f32_16x16x32_bf16(false, a, false, bm, (short)0, acc, false, false);
    }
    const int rowOff = (lane >> 4) * 8;
#pragma unroll
    for (int r = 0; r < 8; ++r) {
      const int n = n0 + m16 + rowOff + r;
      const int c = h * 64 + n16 + (lane & 15);
      const float v = acc[r];
      const float g = 0.5f * v * (1.0f + erff(v * 0.70710678118654752f));  // exact GELU
      attn[((long)b * 4096 + n) * 512 + c] = (bf16)g;
    }
  }
}

// ---------------------------------------------------------------------------
// Kernel 9: final channel LayerNorm, fp32 in place on d_out
// ---------------------------------------------------------------------------
__global__ __launch_bounds__(256)
void ln_inplace_kernel(float* __restrict__ y, const float* __restrict__ g)
{
  const int n = blockIdx.x * 256 + threadIdx.x;
  const int b = blockIdx.y;
  float* yp = y + (long)b * 512 * 4096 + n;
  float s = 0.f, ss = 0.f;
  for (int c = 0; c < 512; ++c) { float v = yp[(long)c * 4096]; s += v; ss += v * v; }
  const float mean = s * (1.0f / 512.0f);
  const float rstd = rsqrtf(ss * (1.0f / 512.0f) - mean * mean + 1e-5f);
  for (int c = 0; c < 512; ++c) {
    float v = yp[(long)c * 4096];
    yp[(long)c * 4096] = (v - mean) * rstd * g[c];
  }
}

// ---------------------------------------------------------------------------
extern "C" void kernel_launch(void* const* d_in, const int* in_sizes, int n_in,
                              void* d_out, int out_size, void* d_ws, size_t ws_size,
                              hipStream_t stream)
{
  const float* fmap  = (const float*)d_in[0];  // [16,512,64,64]
  const float* g1    = (const float*)d_in[1];  // [512]
  const float* w_qkv = (const float*)d_in[2];  // [1536,512]
  const float* w_out = (const float*)d_in[3];  // [512,512]
  const float* g2    = (const float*)d_in[4];  // [512]

  char* ws = (char*)d_ws;
  bf16* x_bf  = (bf16*)ws;                                   // Xt [16][4096][512] (reused as attn)
  bf16* qkv   = (bf16*)(ws + 67108864L);                     // [16][1536][4096]
  bf16* ctxT  = (bf16*)(ws + 67108864L + 201326592L);        // [16][8][64][64]
  bf16* wq_bf = (bf16*)(ws + 67108864L + 201326592L + 1048576L);            // [1536][512]
  bf16* wo_bf = (bf16*)(ws + 67108864L + 201326592L + 1048576L + 1572864L); // [512][512]
  float* out  = (float*)d_out;                               // [16,512,64,64] fp32

  // 0. weights -> bf16 (once per launch; tiny)
  f32_to_bf16_kernel<<<dim3((1536 * 512) / 256), 256, 0, stream>>>(w_qkv, wq_bf, 1536 * 512);
  f32_to_bf16_kernel<<<dim3((512 * 512) / 256), 256, 0, stream>>>(w_out, wo_bf, 512 * 512);
  // 1. LN + bf16 convert, transposed to [n][c]
  ln_bf16_t_kernel<<<dim3(16, 16), 256, 0, stream>>>(fmap, g1, x_bf);
  // 2. QKV GEMM (bf16 WMMA + async-LDS double buffering)
  gemm_wmma_kernel<true><<<dim3(64, 12, 16), 256, 0, stream>>>(
      wq_bf, x_bf, qkv, 1536, 512, 4096, 4096L * 512, 1536L * 4096);
  // 3-5. activation normalizations (in-place bf16)
  q_softmax_kernel<<<dim3(16, 8, 16), 256, 0, stream>>>(qkv);
  k_softmax_kernel<<<dim3(16 * 512), 256, 0, stream>>>(qkv);
  v_l2norm_kernel<<<dim3(16, 8, 16), 256, 0, stream>>>(qkv);
  // 6. context^T per head (bf16 WMMA, K=4096, global-fed fragments)
  context_kernel<<<dim3(128), 512, 0, stream>>>(qkv, ctxT);
  // 7. out = q' x context + GELU, written [n][c] into x_bf
  attn_out_kernel<<<dim3(64, 128), 256, 0, stream>>>(qkv, ctxT, x_bf);
  // 8. out-projection GEMM -> fp32 d_out ([o][n] layout)
  gemm_wmma_kernel<false><<<dim3(64, 4, 16), 256, 0, stream>>>(
      wo_bf, x_bf, (void*)out, 512, 512, 4096, 4096L * 512, 512L * 4096);
  // 9. final LN in place
  ln_inplace_kernel<<<dim3(16, 16), 256, 0, stream>>>(out, g2);
}
